// RelativeMultiHeadAttention_7988639171242
// MI455X (gfx1250) — compile-verified, matrix operand order checked by reading, not run on hardware
//
#include <hip/hip_runtime.h>
#include <hip/hip_bf16.h>
#include <stdint.h>

typedef __attribute__((ext_vector_type(16))) __bf16   v16bf;
typedef __attribute__((ext_vector_type(8)))  float    v8f;
typedef __attribute__((ext_vector_type(4)))  uint32_t u32x4;
typedef __attribute__((ext_vector_type(8)))  int      i32x8;
typedef __attribute__((ext_vector_type(4)))  int      i32x4;

#define BB 4
#define SS 2048
#define DD 1024
#define HH 16
#define DH 64

#if __has_builtin(__builtin_amdgcn_tensor_load_to_lds) && \
    __has_builtin(__builtin_amdgcn_s_wait_tensorcnt)
#define HAVE_TDM 1
#else
#define HAVE_TDM 0
#endif

__device__ __forceinline__ v8f wmma_bf16(v16bf a, v16bf b, v8f c) {
  return __builtin_amdgcn_wmma_f32_16x16x32_bf16(false, a, false, b, (short)0, c,
                                                 false, false);
}

// 16x32 bf16 WMMA fragment load (A or B operand).
// lane&15 selects the M/N line; half-wave selects K sub-block:
// lanes 0-15 K={0..7,16..23}, lanes 16-31 K={8..15,24..31}.
__device__ __forceinline__ v16bf load_frag(const __bf16* __restrict__ p,
                                           int sK, int sN, int lane) {
  const int n  = lane & 15;
  const int kb = (lane & 16) >> 1;  // 0 or 8
  const __bf16* q = p + n * sN;
  v16bf f;
#pragma unroll
  for (int e = 0; e < 8; ++e) f[e]     = q[(kb + e) * sK];
#pragma unroll
  for (int e = 0; e < 8; ++e) f[e + 8] = q[(16 + kb + e) * sK];
  return f;
}

// ---------------- cross-lane reductions within 16-lane groups ---------------
#if __has_builtin(__builtin_amdgcn_permlane16)
__device__ __forceinline__ float pl16(float x, unsigned s0, unsigned s1) {
  int xi = __float_as_int(x);
  xi = __builtin_amdgcn_permlane16(xi, xi, (int)s0, (int)s1, false, false);
  return __int_as_float(xi);
}
__device__ __forceinline__ float fmax16(float x) {
  x = fmaxf(x, pl16(x, 0x67452301u, 0xEFCDAB89u));  // xor 1
  x = fmaxf(x, pl16(x, 0x45670123u, 0xCDEF89ABu));  // xor 3
  x = fmaxf(x, pl16(x, 0x32107654u, 0xBA98FEDCu));  // xor 4
  x = fmaxf(x, pl16(x, 0xFEDCBA98u, 0x76543210u));  // xor 8
  return x;
}
__device__ __forceinline__ float fsum16(float x) {
  x += pl16(x, 0x67452301u, 0xEFCDAB89u);
  x += pl16(x, 0x45670123u, 0xCDEF89ABu);
  x += pl16(x, 0x32107654u, 0xBA98FEDCu);
  x += pl16(x, 0xFEDCBA98u, 0x76543210u);
  return x;
}
#else
template <int PAT>
__device__ __forceinline__ float swz(float x) {
  return __int_as_float(__builtin_amdgcn_ds_swizzle(__float_as_int(x), PAT));
}
__device__ __forceinline__ float fmax16(float x) {
  x = fmaxf(x, swz<0x041f>(x));
  x = fmaxf(x, swz<0x081f>(x));
  x = fmaxf(x, swz<0x101f>(x));
  x = fmaxf(x, swz<0x201f>(x));
  return x;
}
__device__ __forceinline__ float fsum16(float x) {
  x += swz<0x041f>(x);
  x += swz<0x081f>(x);
  x += swz<0x101f>(x);
  x += swz<0x201f>(x);
  return x;
}
#endif

// ------------------------- Tensor Data Mover helper -------------------------
#if HAVE_TDM
// 2D tile (tileW x tileH, bf16) global->LDS with per-row LDS padding.
// padInterval: pad after 2^(v+1) dwords; padAmount: (v+1) dwords of pad.
__device__ __forceinline__ void tdm_load_2d(uint32_t ldsOff, const void* g,
                                            uint32_t tileW, uint32_t tileH,
                                            uint32_t strideElems,
                                            uint32_t padInterval,
                                            uint32_t padAmount) {
  const uint64_t ga = (uint64_t)(uintptr_t)g;
  u32x4 g0;
  g0[0] = 1u;                                  // count=1, user mode
  g0[1] = ldsOff;                              // lds_addr (bytes)
  g0[2] = (uint32_t)ga;                        // global_addr[31:0]
  g0[3] = (uint32_t)((ga >> 32) & 0x01FFFFFFu) | (2u << 30);  // [56:32]|type=2
  const uint32_t td0 = 1u << 30;               // huge tensor dims: no clipping
  const uint32_t td1 = 1u << 30;
  i32x8 g1;
  g1[0] = (int)((1u << 16) |                   // data_size = 2 bytes
                (1u << 20) |                   // pad_enable
                (padInterval << 22) | (padAmount << 25));
  g1[1] = (int)((td0 & 0xFFFFu) << 16);
  g1[2] = (int)(((td0 >> 16) & 0xFFFFu) | ((td1 & 0xFFFFu) << 16));
  g1[3] = (int)(((td1 >> 16) & 0xFFFFu) | ((tileW & 0xFFFFu) << 16));
  g1[4] = (int)(tileH & 0xFFFFu);
  g1[5] = (int)strideElems;                    // tensor_dim0_stride[31:0]
  g1[6] = 0;
  g1[7] = 0;
  const i32x4 z4 = {0, 0, 0, 0};
  const i32x8 z8 = {0, 0, 0, 0, 0, 0, 0, 0};
  __builtin_amdgcn_tensor_load_to_lds(g0, g1, z4, z4, z8, 0);
}
#endif

// ---------------------------------------------------------------------------
// f32 -> bf16 conversion (4 elems/thread)
// ---------------------------------------------------------------------------
__global__ __launch_bounds__(256) void cvt_bf16(const float* __restrict__ in,
                                                __bf16* __restrict__ out,
                                                int n4) {
  const int i = blockIdx.x * blockDim.x + threadIdx.x;
  if (i < n4) {
    const float4 f = ((const float4*)in)[i];
    out[4 * i + 0] = (__bf16)f.x;
    out[4 * i + 1] = (__bf16)f.y;
    out[4 * i + 2] = (__bf16)f.z;
    out[4 * i + 3] = (__bf16)f.w;
  }
}

// ---------------------------------------------------------------------------
// Tiled GEMM: C[M,N] = A[M,K] * W[N,K]^T + bias[N]; A,W bf16.
// 256 thr / 8 waves, 128x128 tile, K-step 32. TDM double-buffered staging.
// ---------------------------------------------------------------------------
template <typename TOUT>
__global__ __launch_bounds__(256) void gemm_bias_wmma(
    const __bf16* __restrict__ A, const __bf16* __restrict__ W,
    const float* __restrict__ bias, TOUT* __restrict__ C, int M, int N, int K) {
  __shared__ __bf16 lA[2][128][40];   // 40 = 32 + 4-dword pad (16B aligned)
  __shared__ __bf16 lB[2][128][40];
  const int tid  = threadIdx.x;
  const int lane = tid & 31;
  const int wave = tid >> 5;
  const int m0 = blockIdx.y * 128;
  const int n0 = blockIdx.x * 128;

  v8f acc[8];
#pragma unroll
  for (int n = 0; n < 8; ++n) acc[n] = (v8f){};

  auto compute = [&](const __bf16* pA, const __bf16* pB) {
    const v16bf af = load_frag(pA + wave * 16 * 40, 1, 40, lane);
#pragma unroll
    for (int n = 0; n < 8; ++n) {
      const v16bf bfr = load_frag(pB + n * 16 * 40, 1, 40, lane);
      acc[n] = wmma_bf16(af, bfr, acc[n]);
    }
  };

  const int nIter = K / 32;
#if HAVE_TDM
  if (wave == 0) {
    tdm_load_2d((uint32_t)(uintptr_t)&lA[0][0][0], A + (size_t)m0 * K,
                32, 128, (uint32_t)K, 3, 3);
    tdm_load_2d((uint32_t)(uintptr_t)&lB[0][0][0], W + (size_t)n0 * K,
                32, 128, (uint32_t)K, 3, 3);
  }
  for (int i = 0; i < nIter; ++i) {
    __syncthreads();  // consumers of the buffer we are about to refill are done
    if (wave == 0) {
      if (i + 1 < nIter) {
        const int kn = (i + 1) * 32;
        const int nb = (i + 1) & 1;
        tdm_load_2d((uint32_t)(uintptr_t)&lA[nb][0][0],
                    A + (size_t)m0 * K + kn, 32, 128, (uint32_t)K, 3, 3);
        tdm_load_2d((uint32_t)(uintptr_t)&lB[nb][0][0],
                    W + (size_t)n0 * K + kn, 32, 128, (uint32_t)K, 3, 3);
        __builtin_amdgcn_s_wait_tensorcnt(2);  // current buffer pair complete
      } else {
        __builtin_amdgcn_s_wait_tensorcnt(0);
      }
    }
    __syncthreads();
    compute(&lA[i & 1][0][0], &lB[i & 1][0][0]);
  }
#else
  for (int i = 0; i < nIter; ++i) {
    const int k0 = i * 32;
    __syncthreads();
    {
      const int r  = tid >> 1;
      const int cb = (tid & 1) * 16;
      const __bf16* sa = A + (size_t)(m0 + r) * K + k0 + cb;
      const __bf16* sw = W + (size_t)(n0 + r) * K + k0 + cb;
#pragma unroll
      for (int e = 0; e < 16; ++e) lA[0][r][cb + e] = sa[e];
#pragma unroll
      for (int e = 0; e < 16; ++e) lB[0][r][cb + e] = sw[e];
    }
    __syncthreads();
    compute(&lA[0][0][0], &lB[0][0][0]);
  }
#endif

  // Epilogue: C layout -> lane&15 = column, VGPR v = row v (+8 in high half)
  const int col   = lane & 15;
  const int rbase = (lane & 16) ? 8 : 0;
#pragma unroll
  for (int n = 0; n < 8; ++n) {
    const int gn = n0 + n * 16 + col;
    const float bv = bias[gn];
#pragma unroll
    for (int v = 0; v < 8; ++v) {
      const int gm = m0 + wave * 16 + rbase + v;
      C[(size_t)gm * N + gn] = (TOUT)(acc[n][v] + bv);
    }
  }
}

// ---------------------------------------------------------------------------
// RoPE in place on (B,S,H,dh) bf16. posIsHead=1: pos=head (reference quirk, q)
// posIsHead=0: pos=time (k).
// ---------------------------------------------------------------------------
__global__ __launch_bounds__(256) void rope_kernel(__bf16* __restrict__ x,
                                                   int posIsHead) {
  const int i  = blockIdx.x * blockDim.x + threadIdx.x;  // pair id
  const int pr = i & 31;
  const int h  = (i >> 5) & 15;
  const int s  = (i >> 9) & 2047;
  const int pos = posIsHead ? h : s;
  const float inv = __expf(-(float)(2 * pr) * (9.2103403719761836f / 64.0f));
  const float ang = (float)pos * inv;
  float sn, cs;
  __sincosf(ang, &sn, &cs);
  __bf16* p = x + 2 * (size_t)i;
  const float x0 = (float)p[0];
  const float x1 = (float)p[1];
  p[0] = (__bf16)(x0 * cs - x1 * sn);
  p[1] = (__bf16)(x1 * cs + x0 * sn);
}

// ---------------------------------------------------------------------------
// V transpose: (B,S,H,dh) -> Vt (B,H,dh,S). 64x64 LDS tiles.
// ---------------------------------------------------------------------------
__global__ __launch_bounds__(256) void transpose_v(
    const __bf16* __restrict__ Vb, __bf16* __restrict__ Vt) {
  __shared__ __bf16 t[64][72];
  const int b  = blockIdx.z;
  const int h  = blockIdx.y;
  const int s0 = blockIdx.x * 64;
  const int tid = threadIdx.x;
  {
    const int r  = tid >> 2;         // s within tile
    const int cb = (tid & 3) * 16;   // d
    const __bf16* src = Vb + ((size_t)(b * SS + s0 + r) * HH + h) * DH + cb;
#pragma unroll
    for (int e = 0; e < 16; ++e) t[r][cb + e] = src[e];
  }
  __syncthreads();
  {
    const int r  = tid >> 2;         // d
    const int cb = (tid & 3) * 16;   // s within tile
    __bf16* dst = Vt + ((size_t)(b * HH + h) * DH + r) * SS + s0 + cb;
#pragma unroll
    for (int e = 0; e < 16; ++e) dst[e] = t[cb + e][r];
  }
}

// ---------------------------------------------------------------------------
// Flash attention. Grid (S/128, H, B), 256 thr (8 waves). 64-key tiles,
// TDM double-buffered. Q,K,Ctx layout (B,S,H,dh); V consumed as Vt (B,H,dh,S)
// so all PV fragments are contiguous b128 LDS loads.
// ---------------------------------------------------------------------------
__global__ __launch_bounds__(256) void attn_wmma(
    const __bf16* __restrict__ Q, const __bf16* __restrict__ Kb,
    const __bf16* __restrict__ Vt, __bf16* __restrict__ Ctx) {
  __shared__ __bf16 lK[2][64][72];   // 72 = 64 + 4-dword pad
  __shared__ __bf16 lVt[2][64][72];  // [d][key]
  __shared__ __bf16 lP[8][16][72];

  const int tid  = threadIdx.x;
  const int lane = tid & 31;
  const int wave = tid >> 5;
  const int b  = blockIdx.z;
  const int h  = blockIdx.y;
  const int q0 = blockIdx.x * 128 + wave * 16;
  const float scale = 0.03125f;  // 1/sqrt(1024)

  // Q fragments (two 16x32 A operands covering dh=64)
  const int n   = lane & 15;
  const int kb8 = (lane & 16) >> 1;
  v16bf qf[2];
#pragma unroll
  for (int c = 0; c < 2; ++c) {
    const __bf16* p = Q + ((size_t)(b * SS + q0 + n) * HH + h) * DH + c * 32;
#pragma unroll
    for (int e = 0; e < 8; ++e) qf[c][e]     = p[kb8 + e];
#pragma unroll
    for (int e = 0; e < 8; ++e) qf[c][e + 8] = p[16 + kb8 + e];
  }

  float mI[8], lI[8];
#pragma unroll
  for (int v = 0; v < 8; ++v) { mI[v] = -1e30f; lI[v] = 0.0f; }
  v8f oacc[4];
#pragma unroll
  for (int v = 0; v < 4; ++v) oacc[v] = (v8f){};

  const int nIter = SS / 64;
#if HAVE_TDM
  auto issue = [&](int s0, int buf) {
    const __bf16* kp = Kb + ((size_t)(b * SS + s0) * HH + h) * DH;
    const __bf16* vp = Vt + ((size_t)(b * HH + h) * DH) * SS + s0;
    tdm_load_2d((uint32_t)(uintptr_t)&lK[buf][0][0], kp, 64, 64, HH * DH, 4, 3);
    tdm_load_2d((uint32_t)(uintptr_t)&lVt[buf][0][0], vp, 64, 64, SS, 4, 3);
  };
  if (wave == 0) issue(0, 0);
#endif
  for (int i = 0; i < nIter; ++i) {
    __syncthreads();  // buffer being refilled is no longer read by anyone
#if HAVE_TDM
    if (wave == 0) {
      if (i + 1 < nIter) {
        issue((i + 1) * 64, (i + 1) & 1);
        __builtin_amdgcn_s_wait_tensorcnt(2);  // current buffer pair complete
      } else {
        __builtin_amdgcn_s_wait_tensorcnt(0);
      }
    }
#else
    {
      const int s0 = i * 64;
      const int r  = tid >> 2;
      const int cb = (tid & 3) * 16;
      const __bf16* kp = Kb + ((size_t)(b * SS + s0 + r) * HH + h) * DH + cb;
      const __bf16* vp = Vt + ((size_t)(b * HH + h) * DH + r) * SS + s0 + cb;
#pragma unroll
      for (int e = 0; e < 16; ++e) lK[0][r][cb + e] = kp[e];
#pragma unroll
      for (int e = 0; e < 16; ++e) lVt[0][r][cb + e] = vp[e];
    }
#endif
    __syncthreads();
#if HAVE_TDM
    const __bf16 (*cK)[72]  = lK[i & 1];
    const __bf16 (*cV)[72]  = lVt[i & 1];
#else
    const __bf16 (*cK)[72]  = lK[0];
    const __bf16 (*cV)[72]  = lVt[0];
#endif

    // scores: four 16x16 key tiles, dh accumulated over 2 WMMA each
    v8f sc[4];
#pragma unroll
    for (int t = 0; t < 4; ++t) {
      v8f s = (v8f){};
      const v16bf b0 = load_frag(&cK[t * 16][0], 1, 72, lane);
      s = wmma_bf16(qf[0], b0, s);
      const v16bf b1 = load_frag(&cK[t * 16][32], 1, 72, lane);
      s = wmma_bf16(qf[1], b1, s);
      sc[t] = s;
    }

    // online softmax per row
#pragma unroll
    for (int v = 0; v < 8; ++v) {
      float a[4];
#pragma unroll
      for (int t = 0; t < 4; ++t) a[t] = sc[t][v] * scale;
      float mx = fmaxf(fmaxf(a[0], a[1]), fmaxf(a[2], a[3]));
      mx = fmax16(mx);
      const float mnew  = fmaxf(mI[v], mx);
      const float alpha = __expf(mI[v] - mnew);
      float p[4], rs = 0.0f;
#pragma unroll
      for (int t = 0; t < 4; ++t) { p[t] = __expf(a[t] - mnew); rs += p[t]; }
      rs = fsum16(rs);
      lI[v] = lI[v] * alpha + rs;
      mI[v] = mnew;
#pragma unroll
      for (int nn = 0; nn < 4; ++nn) oacc[nn][v] *= alpha;
      const int row = (lane & 16) ? v + 8 : v;
#pragma unroll
      for (int t = 0; t < 4; ++t) lP[wave][row][t * 16 + n] = (__bf16)p[t];
    }
    __syncthreads();

    // O += P(16x64) x V(64x64): all fragments contiguous (b128 LDS loads)
    const v16bf pf0 = load_frag(&lP[wave][0][0], 1, 72, lane);
    const v16bf pf1 = load_frag(&lP[wave][0][32], 1, 72, lane);
#pragma unroll
    for (int nn = 0; nn < 4; ++nn) {
      const v16bf vf0 = load_frag(&cV[nn * 16][0], 1, 72, lane);
      oacc[nn] = wmma_bf16(pf0, vf0, oacc[nn]);
      const v16bf vf1 = load_frag(&cV[nn * 16][32], 1, 72, lane);
      oacc[nn] = wmma_bf16(pf1, vf1, oacc[nn]);
    }
  }

  // normalize + store ctx (B,S,H,dh)
  const int col   = lane & 15;
  const int rbase = (lane & 16) ? 8 : 0;
#pragma unroll
  for (int nn = 0; nn < 4; ++nn) {
#pragma unroll
    for (int v = 0; v < 8; ++v) {
      const int r = rbase + v;
      const float o = oacc[nn][v] / lI[v];
      Ctx[((size_t)(b * SS + q0 + r) * HH + h) * DH + nn * 16 + col] =
          (__bf16)o;
    }
  }
}

// ---------------------------------------------------------------------------
extern "C" void kernel_launch(void* const* d_in, const int* in_sizes, int n_in,
                              void* d_out, int out_size, void* d_ws,
                              size_t ws_size, hipStream_t stream) {
  const float* query = (const float*)d_in[0];
  const float* key   = (const float*)d_in[1];
  const float* value = (const float*)d_in[2];
  const float* Wq = (const float*)d_in[3];
  const float* bq = (const float*)d_in[4];
  const float* Wk = (const float*)d_in[5];
  const float* bk = (const float*)d_in[6];
  const float* Wv = (const float*)d_in[7];
  const float* bv = (const float*)d_in[8];
  const float* Wo = (const float*)d_in[9];
  const float* bo = (const float*)d_in[10];
  float* out = (float*)d_out;

  const size_t TB = (size_t)BB * SS * DD * sizeof(__bf16);  // 16 MiB
  const size_t WBy = (size_t)DD * DD * sizeof(__bf16);      // 2 MiB
  char* ws = (char*)d_ws;
  __bf16* qb  = (__bf16*)(ws + 0 * TB);
  __bf16* kb  = (__bf16*)(ws + 1 * TB);
  __bf16* vb  = (__bf16*)(ws + 2 * TB);
  __bf16* cb  = (__bf16*)(ws + 3 * TB);
  __bf16* qx  = (__bf16*)(ws + 4 * TB);
  __bf16* kx  = (__bf16*)(ws + 5 * TB);
  __bf16* vx  = (__bf16*)(ws + 6 * TB);
  __bf16* wqx = (__bf16*)(ws + 7 * TB);
  __bf16* wkx = (__bf16*)(ws + 7 * TB + 1 * WBy);
  __bf16* wvx = (__bf16*)(ws + 7 * TB + 2 * WBy);
  __bf16* wox = (__bf16*)(ws + 7 * TB + 3 * WBy);
  __bf16* vt  = (__bf16*)(ws + 7 * TB + 4 * WBy);           // Vt (B,H,dh,S)

  const int M = BB * SS;            // 8192
  const int nAct4 = M * DD / 4;     // 2,097,152
  const int nW4   = DD * DD / 4;    // 262,144

  cvt_bf16<<<nAct4 / 256, 256, 0, stream>>>(query, qx, nAct4);
  cvt_bf16<<<nAct4 / 256, 256, 0, stream>>>(key,   kx, nAct4);
  cvt_bf16<<<nAct4 / 256, 256, 0, stream>>>(value, vx, nAct4);
  cvt_bf16<<<nW4 / 256, 256, 0, stream>>>(Wq, wqx, nW4);
  cvt_bf16<<<nW4 / 256, 256, 0, stream>>>(Wk, wkx, nW4);
  cvt_bf16<<<nW4 / 256, 256, 0, stream>>>(Wv, wvx, nW4);
  cvt_bf16<<<nW4 / 256, 256, 0, stream>>>(Wo, wox, nW4);

  dim3 gg(DD / 128, M / 128);
  gemm_bias_wmma<__bf16><<<gg, 256, 0, stream>>>(qx, wqx, bq, qb, M, DD, DD);
  gemm_bias_wmma<__bf16><<<gg, 256, 0, stream>>>(kx, wkx, bk, kb, M, DD, DD);
  gemm_bias_wmma<__bf16><<<gg, 256, 0, stream>>>(vx, wvx, bv, vb, M, DD, DD);

  const int pairs = BB * SS * HH * (DH / 2);
  rope_kernel<<<pairs / 256, 256, 0, stream>>>(qb, 1);
  rope_kernel<<<pairs / 256, 256, 0, stream>>>(kb, 0);

  transpose_v<<<dim3(SS / 64, HH, BB), 256, 0, stream>>>(vb, vt);

  attn_wmma<<<dim3(SS / 128, HH, BB), 256, 0, stream>>>(qb, kb, vt, cb);

  gemm_bias_wmma<float><<<gg, 256, 0, stream>>>(cb, wox, bo, out, M, DD, DD);
}
